// Sinc_conv2D_74217034875648
// MI455X (gfx1250) — matrix operand order repaired and checked by reading, not compiled, not run on hardware
//
#include <hip/hip_runtime.h>

// ---------------------------------------------------------------------------
// SincNet 2D conv (16,1,512,512) -> (16,64,512,512), 15x15 "same" conv.
// Implicit GEMM on V_WMMA_F32_16X16X32_BF16 with bf16 hi/lo error
// compensation (Ah*Bh + Ah*Bl + Al*Bh) for near-fp32 accuracy.
//   M = 64 output channels (4 tiles of 16)
//   K = 15x15 taps embedded as K' = kh*16+kw in [0,256) (8 chunks of 32;
//       taps with kh==15 || kw==15 have exactly-zero filter weights)
//   N = pixels in 16-wide row segments; block = 32x32 pixel tile, 8 waves.
//   Each wave processes PAIRS of N-groups (gx=0 / gx=16, same row) so the
//   group-invariant A-operand LDS loads feed 6 WMMAs each (LDS-BW headroom).
// ---------------------------------------------------------------------------

#define OUT_C 64
#define KS    15
#define PADK  7
#define BATCH 16
#define HH    512
#define WW    512
#define TILE  32
#define LROW  48          // LDS input row stride / padded dim (46 used + pad)

typedef __attribute__((ext_vector_type(16))) __bf16 v16bf;
typedef __attribute__((ext_vector_type(8)))  float  v8f;

union ABReg { unsigned int u[8]; v16bf v; };

__device__ __forceinline__ unsigned short f32_to_bf16_rne(float f) {
    unsigned int u = __float_as_uint(f);
    unsigned int r = u + 0x7FFFu + ((u >> 16) & 1u);
    return (unsigned short)(r >> 16);
}
__device__ __forceinline__ float bf16_bits_to_f32(unsigned short h) {
    return __uint_as_float(((unsigned int)h) << 16);
}

// 1D factor of the separable sinc filter: f * sinc(pi*f*|t-7|) * hamming(t)
// hamming over n = linspace(0,15,15): ham[t] = 0.54 - 0.46*cos(2*pi*t/14)
__device__ __forceinline__ float sinc_tap(float f, int t) {
    float ham = 0.54f - 0.46f * cosf(6.28318530717958647692f * (float)t / 14.0f);
    int d = t - PADK;
    float base;
    if (d == 0) {
        base = 1.0f;
    } else {
        float ad = (float)(d < 0 ? -d : d);
        float arg = 3.14159265358979323846f * f * ad;
        base = sinf(arg) / arg;
    }
    return f * base * ham;
}

// ---------------------------------------------------------------------------
// Prep kernel: bf16 hi/lo filter A-matrices in the exact WMMA A-operand lane
// layout (16-bit A 16x32, contiguous K order rather than the ISA table):
//   lane L: M = L%16, half h = L/16
//   VGPR v holds K-pair base = h*16 + v*2   (contiguous K, same order as B)
// Buffer layout (dwords): A[(((c*4+mt)*2+s)*32 + lane)*8 + v], s=0 hi, s=1 lo.
// Total 16384 dwords = 64 KB in d_ws.
// ---------------------------------------------------------------------------
__global__ __launch_bounds__(256)
void sinc_prep_filters(const float* __restrict__ f1,
                       const float* __restrict__ band,
                       unsigned int* __restrict__ A) {
    int e = blockIdx.x * blockDim.x + threadIdx.x;   // 0..16383
    int v    =  e        & 7;
    int lane = (e >> 3)  & 31;
    int s    = (e >> 8)  & 1;
    int mt   = (e >> 9)  & 3;
    int c    = (e >> 11) & 7;
    int h  = lane >> 4;
    int oc = mt * 16 + (lane & 15);

    float f1v = f1[oc];
    float f2v = f1v + fabsf(band[oc]);

    int kbase = c * 32 + (h << 4) + (v << 1);
    unsigned int packed = 0u;
    #pragma unroll
    for (int j = 0; j < 2; ++j) {
        int K  = kbase + j;
        int kh = K >> 4, kw = K & 15;
        float w = 0.0f;
        if (kh < KS && kw < KS) {
            w = sinc_tap(f2v, kh) * sinc_tap(f2v, kw)
              - sinc_tap(f1v, kh) * sinc_tap(f1v, kw);
        }
        unsigned short hb = f32_to_bf16_rne(w);
        unsigned short bits = (s == 0)
            ? hb
            : f32_to_bf16_rne(w - bf16_bits_to_f32(hb));
        packed |= ((unsigned int)bits) << (16 * j);
    }
    A[e] = packed;
}

// ---------------------------------------------------------------------------
// Main conv kernel. Dynamic LDS: 64 KB A-matrices + 2 x 48x48 bf16 input tile.
// ---------------------------------------------------------------------------
__global__ __launch_bounds__(256)
void sinc_conv2d_wmma(const float* __restrict__ x,
                      const unsigned int* __restrict__ Afilt,
                      float* __restrict__ out) {
    extern __shared__ unsigned char smem[];
    unsigned int*   sA  = (unsigned int*)smem;                          // 16384 dw
    unsigned short* sIH = (unsigned short*)(smem + 65536);              // 48*48
    unsigned short* sIL = (unsigned short*)(smem + 65536 + LROW*LROW*2);

    const int tx0 = blockIdx.x * TILE;
    const int ty0 = blockIdx.y * TILE;
    const int b   = blockIdx.z;
    const int tid = threadIdx.x;

    // --- Stage filter A-matrices: 16384 dwords, 64 per thread ---
    #pragma unroll
    for (int i = 0; i < 64; ++i)
        sA[tid + i * 256] = Afilt[tid + i * 256];

    // --- Stage input tile (46x46 valid, zero-padded to 48x48), bf16 hi/lo ---
    const float* xb = x + (size_t)b * HH * WW;
    #pragma unroll
    for (int i = 0; i < 9; ++i) {
        int idx = tid + i * 256;        // 0..2303
        int r  = idx / LROW;
        int cc = idx - r * LROW;
        float val = 0.0f;
        int gr = ty0 - PADK + r;
        int gc = tx0 - PADK + cc;
        if (r < 46 && cc < 46 && gr >= 0 && gr < HH && gc >= 0 && gc < WW)
            val = xb[gr * WW + gc];
        unsigned short hb = f32_to_bf16_rne(val);
        sIH[idx] = hb;
        sIL[idx] = f32_to_bf16_rne(val - bf16_bits_to_f32(hb));
    }
    __syncthreads();

    const int wv   = tid >> 5;     // wave id 0..7
    const int lane = tid & 31;
    const int h    = lane >> 4;    // K-half for A/B operands
    const int nn   = lane & 15;    // N (pixel) within segment

    #pragma unroll 1
    for (int p = 0; p < 4; ++p) {
        const int gy = wv * 4 + p;     // tile row 0..31; groups (gy,gx=0),(gy,gx=16)

        v8f a0_0 = {}, a0_1 = {}, a0_2 = {}, a0_3 = {};   // gx = 0
        v8f a1_0 = {}, a1_1 = {}, a1_2 = {}, a1_3 = {};   // gx = 16

        #pragma unroll 1
        for (int c = 0; c < 8; ++c) {
            // B operands (32x16 bf16): lane half h -> K = h*16 + 2v(+1),
            // N = lane%16.  K' = c*32 + h*16 + 2v + j  =>  kh = 2c+h (uniform
            // per half), kw = 2v+j: contiguous halfwords in the LDS row.
            ABReg bh0, bl0, bh1, bl1;
            const int base = (gy + 2 * c + h) * LROW + nn;
            #pragma unroll
            for (int v = 0; v < 8; ++v) {
                unsigned int x0h = sIH[base + 2 * v];
                unsigned int x1h = sIH[base + 2 * v + 1];
                bh0.u[v] = x0h | (x1h << 16);
                unsigned int x0l = sIL[base + 2 * v];
                unsigned int x1l = sIL[base + 2 * v + 1];
                bl0.u[v] = x0l | (x1l << 16);
                unsigned int y0h = sIH[base + 16 + 2 * v];
                unsigned int y1h = sIH[base + 16 + 2 * v + 1];
                bh1.u[v] = y0h | (y1h << 16);
                unsigned int y0l = sIL[base + 16 + 2 * v];
                unsigned int y1l = sIL[base + 16 + 2 * v + 1];
                bl1.u[v] = y0l | (y1l << 16);
            }
            #pragma unroll
            for (int mt = 0; mt < 4; ++mt) {
                ABReg ah, al;
                const unsigned int* pah = sA + (((c * 4 + mt) * 2 + 0) * 32 + lane) * 8;
                const unsigned int* pal = sA + (((c * 4 + mt) * 2 + 1) * 32 + lane) * 8;
                #pragma unroll
                for (int v = 0; v < 8; ++v) { ah.u[v] = pah[v]; al.u[v] = pal[v]; }

                v8f* p0 = (mt == 0) ? &a0_0 : (mt == 1) ? &a0_1
                        : (mt == 2) ? &a0_2 : &a0_3;
                v8f* p1 = (mt == 0) ? &a1_0 : (mt == 1) ? &a1_1
                        : (mt == 2) ? &a1_2 : &a1_3;
                // w*x ~= Ah*Bh + Ah*Bl + Al*Bh  (bf16x3, ~fp32 accuracy)
                *p0 = __builtin_amdgcn_wmma_f32_16x16x32_bf16(
                          false, ah.v, false, bh0.v, (short)0, *p0, false, false);
                *p0 = __builtin_amdgcn_wmma_f32_16x16x32_bf16(
                          false, ah.v, false, bl0.v, (short)0, *p0, false, false);
                *p0 = __builtin_amdgcn_wmma_f32_16x16x32_bf16(
                          false, al.v, false, bh0.v, (short)0, *p0, false, false);
                *p1 = __builtin_amdgcn_wmma_f32_16x16x32_bf16(
                          false, ah.v, false, bh1.v, (short)0, *p1, false, false);
                *p1 = __builtin_amdgcn_wmma_f32_16x16x32_bf16(
                          false, ah.v, false, bl1.v, (short)0, *p1, false, false);
                *p1 = __builtin_amdgcn_wmma_f32_16x16x32_bf16(
                          false, al.v, false, bh1.v, (short)0, *p1, false, false);
            }
        }

        // --- Store D: VGPR v, lane half h -> oc = mt*16 + h*8 + v; N = nn ---
        const int y = ty0 + gy;
        #pragma unroll
        for (int mt = 0; mt < 4; ++mt) {
            const v8f d0 = (mt == 0) ? a0_0 : (mt == 1) ? a0_1
                         : (mt == 2) ? a0_2 : a0_3;
            const v8f d1 = (mt == 0) ? a1_0 : (mt == 1) ? a1_1
                         : (mt == 2) ? a1_2 : a1_3;
            #pragma unroll
            for (int v = 0; v < 8; ++v) {
                int oc = mt * 16 + (h << 3) + v;
                float* prow = out + (((size_t)(b * OUT_C + oc) * HH + y) * WW);
                __builtin_nontemporal_store(d0[v], prow + tx0 + nn);        // 1.07 GB
                __builtin_nontemporal_store(d1[v], prow + tx0 + 16 + nn);   // streaming
            }
        }
    }
}

extern "C" void kernel_launch(void* const* d_in, const int* in_sizes, int n_in,
                              void* d_out, int out_size, void* d_ws, size_t ws_size,
                              hipStream_t stream) {
    const float* x    = (const float*)d_in[0];   // (16,1,512,512) f32
    const float* f1   = (const float*)d_in[1];   // (64,) f32
    const float* band = (const float*)d_in[2];   // (64,) f32
    // d_in[3] = kernel_size (15), compile-time constant here
    unsigned int* A = (unsigned int*)d_ws;       // 64 KB of filter A-matrices
    float* out = (float*)d_out;

    sinc_prep_filters<<<64, 256, 0, stream>>>(f1, band, A);

    dim3 grid(WW / TILE, HH / TILE, BATCH);      // 16 x 16 x 16 blocks
    size_t shmem = 65536 + 2 * (size_t)(LROW * LROW) * 2;  // 74752 B (< 320 KB WGP)
    sinc_conv2d_wmma<<<grid, 256, shmem, stream>>>(x, A, out);
}
